// DMAN_attention_31267361915241
// MI455X (gfx1250) — matrix operand order repaired and checked by reading, not compiled, not run on hardware
//
#include <hip/hip_runtime.h>
#include <stdint.h>

#define T_SEQ   1024
#define BATCH   4
#define EMB     1024
#define NHEAD   16
#define HD      64
#define MAXLEN  1024
#define PT_LEN  (2*MAXLEN + 1)   // 2049

typedef __attribute__((ext_vector_type(16))) __bf16 v16bf;
typedef __attribute__((ext_vector_type(8)))  float  v8f;

union Frag16 {
    v16bf v;
    uint4 q[2];
    unsigned short s[16];
};

// Native fptrunc (RNE) -> lets the backend use v_cvt(_pk)_bf16_f32.
__device__ __forceinline__ unsigned short f2bf(float f) {
    __bf16 h = (__bf16)f;
    return __builtin_bit_cast(unsigned short, h);
}
// Pack two converts into one dword (encourages v_cvt_pk + b32 stores).
__device__ __forceinline__ uint32_t f2bf2(float flo, float fhi) {
    return (uint32_t)f2bf(flo) | ((uint32_t)f2bf(fhi) << 16);
}

// ---------------------------------------------------------------------------
// Kernel 1: ctx[b][t] = query[t,b,:] . ctx_w  (one wave per row)
// ---------------------------------------------------------------------------
__global__ __launch_bounds__(256) void ctx_kernel(
    const float* __restrict__ query, const float* __restrict__ ctx_w,
    float* __restrict__ ctx_ws)                     // ctx_ws[b*T + t]
{
    const int row  = blockIdx.x * 8 + (threadIdx.x >> 5);   // t*B + b
    const int lane = threadIdx.x & 31;
    const float* qp = query + (size_t)row * EMB;
    float s = 0.f;
    #pragma unroll
    for (int i = 0; i < 8; ++i) {
        const int e = (i * 32 + lane) * 4;
        const float4 a = *(const float4*)(qp + e);
        const float4 c = *(const float4*)(ctx_w + e);
        s += a.x*c.x + a.y*c.y + a.z*c.z + a.w*c.w;
    }
    #pragma unroll
    for (int msk = 16; msk >= 1; msk >>= 1) s += __shfl_xor(s, msk, 32);
    if (lane == 0) {
        const int t = row >> 2, b = row & 3;
        ctx_ws[b * T_SEQ + t] = s;
    }
}

// ---------------------------------------------------------------------------
// Kernel 2: QKV projection. M=4096(T*B), K=1024, N=3072. bf16 WMMA.
// Block 256 thr (8 waves), block tile 64x128, wave tile 32x32 (4 WMMA/k-step).
// Writes q/k head-major, v TRANSPOSED (BH,HD,T) for contiguous P*V fragments.
// ---------------------------------------------------------------------------
__global__ __launch_bounds__(256) void qkv_proj_kernel(
    const float* __restrict__ query,      // (T,B,E) == (M,K)
    const float* __restrict__ w,          // (3E,E)  N-major for B operand
    const float* __restrict__ bias,       // (3E)
    unsigned short* __restrict__ q_ws,    // (BH,T,HD) bf16
    unsigned short* __restrict__ k_ws,    // (BH,T,HD) bf16
    unsigned short* __restrict__ vT_ws)   // (BH,HD,T) bf16
{
    __shared__ __align__(16) unsigned short lA[64][32];
    __shared__ __align__(16) unsigned short lB[128][32];
    const int tid  = threadIdx.x;
    const int lane = tid & 31;
    const int wid  = tid >> 5;
    const int wr   = wid >> 2;           // 0..1 -> +32 rows
    const int wc   = wid & 3;            // 0..3 -> +32 cols
    const int m0   = blockIdx.x * 64;
    const int n0   = blockIdx.y * 128;
    const int lo   = lane & 15, hi = lane >> 4;
    const int srow = tid >> 3;           // 0..31
    const int scol = (tid & 7) * 4;

    v8f acc[2][2] = {};
    for (int k0 = 0; k0 < EMB; k0 += 32) {
        if (k0 + 32 < EMB) {  // warm next chunk (global_prefetch_b8)
            __builtin_prefetch(query + (size_t)(m0 + srow) * EMB + k0 + 32 + scol, 0, 1);
            __builtin_prefetch(w     + (size_t)(n0 + srow) * EMB + k0 + 32 + scol, 0, 1);
        }
        __syncthreads();
        #pragma unroll
        for (int i = 0; i < 2; ++i) {     // A: 64 rows
            const int r = srow + 32*i;
            const float4 f = *(const float4*)(query + (size_t)(m0 + r) * EMB + k0 + scol);
            ((uint32_t*)&lA[r][scol])[0] = f2bf2(f.x, f.y);
            ((uint32_t*)&lA[r][scol])[1] = f2bf2(f.z, f.w);
        }
        #pragma unroll
        for (int i = 0; i < 4; ++i) {     // B: 128 rows (N-major)
            const int r = srow + 32*i;
            const float4 g = *(const float4*)(w + (size_t)(n0 + r) * EMB + k0 + scol);
            ((uint32_t*)&lB[r][scol])[0] = f2bf2(g.x, g.y);
            ((uint32_t*)&lB[r][scol])[1] = f2bf2(g.z, g.w);
        }
        __syncthreads();
        Frag16 a[2], b[2];
        #pragma unroll
        for (int mi = 0; mi < 2; ++mi) {
            const int am = lo + 32*wr + 16*mi;
            a[mi].q[0] = *(const uint4*)&lA[am][hi * 8];
            a[mi].q[1] = *(const uint4*)&lA[am][hi * 8 + 16];
        }
        #pragma unroll
        for (int ni = 0; ni < 2; ++ni) {
            const int bn = lo + 32*wc + 16*ni;
            b[ni].q[0] = *(const uint4*)&lB[bn][hi * 8];
            b[ni].q[1] = *(const uint4*)&lB[bn][hi * 8 + 16];
        }
        #pragma unroll
        for (int mi = 0; mi < 2; ++mi)
            #pragma unroll
            for (int ni = 0; ni < 2; ++ni)
                acc[mi][ni] = __builtin_amdgcn_wmma_f32_16x16x32_bf16(
                    false, a[mi].v, false, b[ni].v, (short)0, acc[mi][ni], false, false);
    }
    // Epilogue: +bias, scatter to q/k/vT.  `which`/`h` are wave-uniform per
    // fragment; row base is a multiple of 8 => b = r&3, t = tbase + (r>>2).
    #pragma unroll
    for (int mi = 0; mi < 2; ++mi) {
        const int tbase = (m0 + 32*wr + 16*mi + 8*hi) >> 2;
        #pragma unroll
        for (int ni = 0; ni < 2; ++ni) {
            const int n     = n0 + 32*wc + 16*ni + lo;
            const float bs  = bias[n];
            const int which = n >> 10;              // wave-uniform
            const int h     = (n & 1023) >> 6;      // wave-uniform
            const int hd    = n & 63;
            if (which == 2) {
                #pragma unroll
                for (int r = 0; r < 8; ++r) {
                    const int bh = (r & 3) * NHEAD + h;
                    const int t  = tbase + (r >> 2);
                    vT_ws[(size_t)bh * (HD*T_SEQ) + hd * T_SEQ + t] =
                        f2bf(acc[mi][ni][r] + bs);
                }
            } else {
                unsigned short* base = which ? k_ws : q_ws;
                #pragma unroll
                for (int r = 0; r < 8; ++r) {
                    const int bh = (r & 3) * NHEAD + h;
                    const int t  = tbase + (r >> 2);
                    base[(size_t)bh * (T_SEQ*HD) + t * HD + hd] =
                        f2bf(acc[mi][ni][r] + bs);
                }
            }
        }
    }
}

// ---------------------------------------------------------------------------
// Kernel 3: flash attention with fused logsigmoid positional gate.
// Block = 128 thr (4 waves); wave handles one 16-query tile of head blockIdx.x.
// Key tiles of 32; online softmax; mask row-normalization cancels in softmax.
// ---------------------------------------------------------------------------
__global__ __launch_bounds__(128) void attn_kernel(
    const unsigned short* __restrict__ q_ws,
    const unsigned short* __restrict__ k_ws,
    const unsigned short* __restrict__ vT_ws,
    const float* __restrict__ ctx_ws,
    const float* __restrict__ fwd_pos,
    const float* __restrict__ bwd_pos,
    const float* __restrict__ head_x,
    unsigned short* __restrict__ o_ws)     // (T,B,E) bf16
{
    __shared__ __align__(16) float lpos[PT_LEN];
    __shared__ __align__(16) unsigned short lP[4][16][32];  // wave-private P tile

    const int tid  = threadIdx.x;
    const int lane = tid & 31;
    const int wid  = tid >> 5;
    const int bh   = blockIdx.x;
    const int b    = bh >> 4;
    const int h    = bh & 15;
    const int t0   = (blockIdx.y * 4 + wid) * 16;
    const int lo   = lane & 15, hi = lane >> 4;

    // stage pos_table = [fwd(1024), 0, bwd(1024)]
    for (int i = tid; i < PT_LEN; i += 128) {
        float v;
        if (i < MAXLEN)       v = fwd_pos[i];
        else if (i == MAXLEN) v = 0.f;
        else                  v = bwd_pos[i - MAXLEN - 1];
        lpos[i] = v;
    }
    __syncthreads();

    // Q A-fragments: K chunks 0..31 and 32..63 (held for the whole key loop)
    Frag16 aq0, aq1;
    {
        const unsigned short* qrow = q_ws + ((size_t)bh * T_SEQ + t0 + lo) * HD;
        const int kb = hi * 8;
        aq0.q[0] = *(const uint4*)(qrow + kb);
        aq0.q[1] = *(const uint4*)(qrow + kb + 16);
        aq1.q[0] = *(const uint4*)(qrow + 32 + kb);
        aq1.q[1] = *(const uint4*)(qrow + 32 + kb + 16);
    }

    const float hx = head_x[h];
    float z0[8], rmax[8], rsum[8];
    #pragma unroll
    for (int r = 0; r < 8; ++r) {
        z0[r]   = ctx_ws[b * T_SEQ + t0 + r + 8*hi] + hx;
        rmax[r] = -1e30f;
        rsum[r] = 0.f;
    }
    v8f o[4] = {};

    for (int s0 = 0; s0 < T_SEQ; s0 += 32) {
        if (s0 + 32 < T_SEQ) {  // warm next key/value tiles
            __builtin_prefetch(k_ws + ((size_t)bh * T_SEQ + s0 + 32 + lane) * HD, 0, 1);
            __builtin_prefetch(vT_ws + ((size_t)bh * HD + 2*lane) * T_SEQ + s0 + 32, 0, 1);
        }
        // ---- S = Q * K^T  (two 16-wide key subtiles, K=64 in two chunks) ----
        v8f S[2];
        #pragma unroll
        for (int j = 0; j < 2; ++j) {
            Frag16 bk0, bk1;
            const unsigned short* krow =
                k_ws + ((size_t)bh * T_SEQ + s0 + 16*j + lo) * HD;
            const int kb = hi * 8;
            bk0.q[0] = *(const uint4*)(krow + kb);
            bk0.q[1] = *(const uint4*)(krow + kb + 16);
            bk1.q[0] = *(const uint4*)(krow + 32 + kb);
            bk1.q[1] = *(const uint4*)(krow + 32 + kb + 16);
            v8f c = {};
            c = __builtin_amdgcn_wmma_f32_16x16x32_bf16(false, aq0.v, false, bk0.v,
                                                        (short)0, c, false, false);
            c = __builtin_amdgcn_wmma_f32_16x16x32_bf16(false, aq1.v, false, bk1.v,
                                                        (short)0, c, false, false);
            S[j] = c;
        }
        // ---- add logsigmoid(ctx + pos[s-t+1024] + head_x) ----
        #pragma unroll
        for (int j = 0; j < 2; ++j) {
            const int s_idx = s0 + 16*j + lo;
            #pragma unroll
            for (int r = 0; r < 8; ++r) {
                const int t = t0 + r + 8*hi;
                const float z  = z0[r] + lpos[s_idx - t + MAXLEN];
                const float ls = fminf(z, 0.f) - __logf(1.f + __expf(-fabsf(z)));
                S[j][r] += ls;
            }
        }
        // ---- online softmax (rows live across 16-lane halves: xor 1..8) ----
        float p0[8], p1[8], scale[8];
        #pragma unroll
        for (int r = 0; r < 8; ++r) {
            float tm = fmaxf(S[0][r], S[1][r]);
            #pragma unroll
            for (int msk = 1; msk <= 8; msk <<= 1) tm = fmaxf(tm, __shfl_xor(tm, msk, 32));
            const float mn = fmaxf(rmax[r], tm);
            scale[r] = __expf(rmax[r] - mn);
            rmax[r]  = mn;
            p0[r] = __expf(S[0][r] - mn);
            p1[r] = __expf(S[1][r] - mn);
            float ts = p0[r] + p1[r];
            #pragma unroll
            for (int msk = 1; msk <= 8; msk <<= 1) ts += __shfl_xor(ts, msk, 32);
            rsum[r] = rsum[r] * scale[r] + ts;
        }
        #pragma unroll
        for (int nt = 0; nt < 4; ++nt)
            #pragma unroll
            for (int r = 0; r < 8; ++r) o[nt][r] *= scale[r];

        // ---- C-layout P -> A-layout via LDS round trip ----
        __syncthreads();
        #pragma unroll
        for (int r = 0; r < 8; ++r) {
            const int m = r + 8*hi;
            lP[wid][m][lo]      = f2bf(p0[r]);
            lP[wid][m][16 + lo] = f2bf(p1[r]);
        }
        __syncthreads();
        Frag16 ap;
        ap.q[0] = *(const uint4*)&lP[wid][lo][hi * 8];
        ap.q[1] = *(const uint4*)&lP[wid][lo][hi * 8 + 16];

        // ---- O += P * V   (V^T is N-major: contiguous B-fragments) ----
        #pragma unroll
        for (int nt = 0; nt < 4; ++nt) {
            Frag16 bv;
            const unsigned short* vrow =
                vT_ws + ((size_t)bh * HD + 16*nt + lo) * T_SEQ + s0;
            const int kb = hi * 8;
            bv.q[0] = *(const uint4*)(vrow + kb);
            bv.q[1] = *(const uint4*)(vrow + kb + 16);
            o[nt] = __builtin_amdgcn_wmma_f32_16x16x32_bf16(
                false, ap.v, false, bv.v, (short)0, o[nt], false, false);
        }
    }

    // ---- normalize and store O in (T,B,E) bf16 for the output GEMM ----
    #pragma unroll
    for (int nt = 0; nt < 4; ++nt) {
        const int e = h * HD + 16*nt + lo;
        #pragma unroll
        for (int r = 0; r < 8; ++r) {
            const int t = t0 + r + 8*hi;
            o_ws[((size_t)t * BATCH + b) * EMB + e] = f2bf(o[nt][r] / rsum[r]);
        }
    }
}

// ---------------------------------------------------------------------------
// Kernel 4: output projection. M=4096, K=1024, N=1024. A bf16, B fp32->bf16.
// Block tile 64x128, wave tile 32x32 (same scheme as kernel 2).
// ---------------------------------------------------------------------------
__global__ __launch_bounds__(256) void out_proj_kernel(
    const unsigned short* __restrict__ A,   // (4096,1024) bf16 (o_ws)
    const float* __restrict__ w,            // (E,E) N-major
    const float* __restrict__ bias,         // (E)
    float* __restrict__ out)                // (T,B,E) fp32
{
    __shared__ __align__(16) unsigned short lA[64][32];
    __shared__ __align__(16) unsigned short lB[128][32];
    const int tid  = threadIdx.x;
    const int lane = tid & 31;
    const int wid  = tid >> 5;
    const int wr   = wid >> 2;
    const int wc   = wid & 3;
    const int m0   = blockIdx.x * 64;
    const int n0   = blockIdx.y * 128;
    const int lo   = lane & 15, hi = lane >> 4;
    const int srow = tid >> 3;
    const int scol = (tid & 7) * 4;

    v8f acc[2][2] = {};
    for (int k0 = 0; k0 < EMB; k0 += 32) {
        if (k0 + 32 < EMB) {
            __builtin_prefetch(A + (size_t)(m0 + srow) * EMB + k0 + 32 + scol, 0, 1);
            __builtin_prefetch(w + (size_t)(n0 + srow) * EMB + k0 + 32 + scol, 0, 1);
        }
        __syncthreads();
        #pragma unroll
        for (int i = 0; i < 2; ++i) {
            const int r = srow + 32*i;
            *(uint2*)&lA[r][scol] =
                *(const uint2*)(A + (size_t)(m0 + r) * EMB + k0 + scol);
        }
        #pragma unroll
        for (int i = 0; i < 4; ++i) {
            const int r = srow + 32*i;
            const float4 g = *(const float4*)(w + (size_t)(n0 + r) * EMB + k0 + scol);
            ((uint32_t*)&lB[r][scol])[0] = f2bf2(g.x, g.y);
            ((uint32_t*)&lB[r][scol])[1] = f2bf2(g.z, g.w);
        }
        __syncthreads();
        Frag16 a[2], b[2];
        #pragma unroll
        for (int mi = 0; mi < 2; ++mi) {
            const int am = lo + 32*wr + 16*mi;
            a[mi].q[0] = *(const uint4*)&lA[am][hi * 8];
            a[mi].q[1] = *(const uint4*)&lA[am][hi * 8 + 16];
        }
        #pragma unroll
        for (int ni = 0; ni < 2; ++ni) {
            const int bn = lo + 32*wc + 16*ni;
            b[ni].q[0] = *(const uint4*)&lB[bn][hi * 8];
            b[ni].q[1] = *(const uint4*)&lB[bn][hi * 8 + 16];
        }
        #pragma unroll
        for (int mi = 0; mi < 2; ++mi)
            #pragma unroll
            for (int ni = 0; ni < 2; ++ni)
                acc[mi][ni] = __builtin_amdgcn_wmma_f32_16x16x32_bf16(
                    false, a[mi].v, false, b[ni].v, (short)0, acc[mi][ni], false, false);
    }
    #pragma unroll
    for (int mi = 0; mi < 2; ++mi) {
        const int rowb = m0 + 32*wr + 16*mi + 8*hi;
        #pragma unroll
        for (int ni = 0; ni < 2; ++ni) {
            const int n = n0 + 32*wc + 16*ni + lo;
            const float bs = bias[n];
            #pragma unroll
            for (int r = 0; r < 8; ++r)
                out[(size_t)(rowb + r) * EMB + n] = acc[mi][ni][r] + bs;
        }
    }
}

// ---------------------------------------------------------------------------
extern "C" void kernel_launch(void* const* d_in, const int* in_sizes, int n_in,
                              void* d_out, int out_size, void* d_ws, size_t ws_size,
                              hipStream_t stream) {
    const float* query = (const float*)d_in[0];   // (T,B,E)
    const float* in_w  = (const float*)d_in[1];   // (3E,E)
    const float* in_b  = (const float*)d_in[2];   // (3E)
    const float* fwd   = (const float*)d_in[3];   // (1024)
    const float* bwd   = (const float*)d_in[4];   // (1024)
    const float* hx    = (const float*)d_in[5];   // (16)
    const float* ctxw  = (const float*)d_in[6];   // (1,E)
    const float* outw  = (const float*)d_in[7];   // (E,E)
    const float* outb  = (const float*)d_in[8];   // (E)
    float* out = (float*)d_out;

    // workspace layout (needs ~34 MB)
    char* ws = (char*)d_ws;
    unsigned short* q_ws  = (unsigned short*)(ws);                        // 8 MB
    unsigned short* k_ws  = (unsigned short*)(ws + ((size_t)8  << 20));   // 8 MB
    unsigned short* vT_ws = (unsigned short*)(ws + ((size_t)16 << 20));   // 8 MB
    unsigned short* o_ws  = (unsigned short*)(ws + ((size_t)24 << 20));   // 8 MB
    float*          ctx_ws= (float*)         (ws + ((size_t)32 << 20));   // 16 KB

    ctx_kernel     <<<512,          256, 0, stream>>>(query, ctxw, ctx_ws);
    qkv_proj_kernel<<<dim3(64, 24), 256, 0, stream>>>(query, in_w, in_b,
                                                      q_ws, k_ws, vT_ws);
    attn_kernel    <<<dim3(64, 16), 128, 0, stream>>>(q_ws, k_ws, vT_ws, ctx_ws,
                                                      fwd, bwd, hx, o_ws);
    out_proj_kernel<<<dim3(64, 8),  256, 0, stream>>>(o_ws, outw, outb, out);
}
